// AnomalyAttention_71382356459861
// MI455X (gfx1250) — matrix operand order repaired and checked by reading, not compiled, not run on hardware
//
#include <hip/hip_runtime.h>

// Problem constants (from reference): B=8, L=S=1024, H=8, E=64
#define BATCH   8
#define SEQL    1024
#define NHEAD   8
#define EDIM    64
#define ROWSTR  (NHEAD * EDIM)   // 512 floats between consecutive l/s rows
#define CHUNK   64               // keys processed per iteration
#define MTILE   128              // query rows per workgroup (8 waves x 16)
#define THREADS 256
#define VSTR    72               // padded stride (bf16 elems) for transposed V in LDS
#define LOG2E   1.4426950408889634f

typedef __attribute__((ext_vector_type(16))) __bf16 v16bf;
typedef __attribute__((ext_vector_type(8)))  float  v8f;

union BF16x16 { v16bf v; __bf16 h[16]; uint4 q[2]; };
union BF16x8  { __bf16 h[8]; uint4 q; };

__device__ __forceinline__ v16bf lds_pair16(const __bf16* p0, const __bf16* p1) {
  BF16x16 r;
  r.q[0] = *reinterpret_cast<const uint4*>(p0);
  r.q[1] = *reinterpret_cast<const uint4*>(p1);
  return r.v;
}

__device__ __forceinline__ v8f wmma_bf16(v16bf a, v16bf b, v8f c) {
  // D = A(16xK=32) * B(32x16) + C, f32 accumulate
  return __builtin_amdgcn_wmma_f32_16x16x32_bf16(false, a, false, b, (short)0, c,
                                                 false, false);
}

__global__ __launch_bounds__(THREADS)
void anomaly_attn_flash_kernel(const float* __restrict__ Q,
                               const float* __restrict__ K,
                               const float* __restrict__ GV,
                               const float* __restrict__ LV,
                               const float* __restrict__ SIG,
                               float* __restrict__ OutG,
                               float* __restrict__ OutL)
{
  __shared__ __attribute__((aligned(16))) __bf16 Klds[CHUNK * EDIM]; // [s][e] row-major
  __shared__ __attribute__((aligned(16))) __bf16 GVt[EDIM * VSTR];   // [e][s] transposed
  __shared__ __attribute__((aligned(16))) __bf16 LVt[EDIM * VSTR];   // [e][s] transposed

  const int tilesPerSeq = SEQL / MTILE;                // 8
  const int bh   = blockIdx.x / tilesPerSeq;
  const int tile = blockIdx.x % tilesPerSeq;
  const int b = bh / NHEAD;
  const int h = bh % NHEAD;

  const int lane   = threadIdx.x & 31;
  const int wave   = threadIdx.x >> 5;
  const int m      = lane & 15;          // query row within the wave's 16-row tile
  const bool hiHalf = (lane >= 16);
  const int rowBase = tile * MTILE + wave * 16;
  const int row     = rowBase + m;       // this lane's query row (global within L)

  const long headBase = (((long)b * SEQL) * NHEAD + h) * EDIM;

  // ---------------- Q as WMMA B-operands (held in VGPRs all kernel) ----------
  // B(32x16) layout: lanes 0-15 hold K(=e)=0..15 of column m, lanes 16-31 hold e=16..31
  const float* qrow = Q + headBase + (long)row * ROWSTR;
  BF16x16 qb0u, qb1u;
  {
    const int e0 = hiHalf ? 16 : 0;
    const float4* p0 = reinterpret_cast<const float4*>(qrow + e0);        // e0..e0+15
    const float4* p1 = reinterpret_cast<const float4*>(qrow + 32 + e0);   // 32+e0..
    float4 a0 = p0[0], a1 = p0[1], a2 = p0[2], a3 = p0[3];
    float4 b0 = p1[0], b1 = p1[1], b2 = p1[2], b3 = p1[3];
    const float qa[16] = {a0.x,a0.y,a0.z,a0.w, a1.x,a1.y,a1.z,a1.w,
                          a2.x,a2.y,a2.z,a2.w, a3.x,a3.y,a3.z,a3.w};
    const float qb[16] = {b0.x,b0.y,b0.z,b0.w, b1.x,b1.y,b1.z,b1.w,
                          b2.x,b2.y,b2.z,b2.w, b3.x,b3.y,b3.z,b3.w};
#pragma unroll
    for (int i = 0; i < 16; ++i) { qb0u.h[i] = (__bf16)qa[i]; qb1u.h[i] = (__bf16)qb[i]; }
  }
  const v16bf qb0 = qb0u.v, qb1 = qb1u.v;

  // ------------- sigma -> Gaussian prior coefficients (log2 domain) ----------
  float sigraw = SIG[((long)b * SEQL + row) * NHEAD + h];
  float sgm = 1.0f / (1.0f + exp2f(-7.2134752044448170f * sigraw)) + 1e-5f; // sigmoid(5x)
  float sig = exp2f(sgm * 1.5849625007211562f) - 1.0f;                      // 3^sgm - 1
  const float gA2 = (0.3989422804014327f * LOG2E) / sig;  // (1/sqrt(2pi)/sig)*log2e
  const float gBn = -(0.5f * LOG2E) / (sig * sig);        // -log2e/(2 sig^2)

  // ---------------- flash-softmax state (log2-scaled scores) -----------------
  const v8f vzero = {0.f,0.f,0.f,0.f,0.f,0.f,0.f,0.f};
  v8f og[4], ol[4];
#pragma unroll
  for (int j = 0; j < 4; ++j) { og[j] = vzero; ol[j] = vzero; }
  float mg = -1e30f, ml = -1e30f;      // running row maxima (global / local)
  float sumg = 0.f, suml = 0.f;        // running row sums

  // staging: each thread converts 8 consecutive floats of rows srow, srow+32
  const int srow = threadIdx.x >> 3;           // 0..31
  const int se0  = (threadIdx.x & 7) * 8;      // 0..56
  const long stagOff = headBase + (long)srow * ROWSTR + se0;

  const int aeb = hiHalf ? 16 : 0;     // K-run base for 16-bit A operands
  const int sbh = hiHalf ? 8 : 0;      // s base for PV B operands

  for (int sc = 0; sc < SEQL; sc += CHUNK) {
    // ======== cooperative staging: K row-major bf16, V arrays transposed =====
#pragma unroll
    for (int half = 0; half < 2; ++half) {
      const int rr = srow + 32 * half;
      const long off = stagOff + (long)(sc + 32 * half) * ROWSTR;
      const float* kp = K  + off;
      const float* gp = GV + off;
      const float* lp = LV + off;
      float4 k0 = *reinterpret_cast<const float4*>(kp);
      float4 k1 = *reinterpret_cast<const float4*>(kp + 4);
      float4 g0 = *reinterpret_cast<const float4*>(gp);
      float4 g1 = *reinterpret_cast<const float4*>(gp + 4);
      float4 l0 = *reinterpret_cast<const float4*>(lp);
      float4 l1 = *reinterpret_cast<const float4*>(lp + 4);

      BF16x8 kb;
      kb.h[0]=(__bf16)k0.x; kb.h[1]=(__bf16)k0.y; kb.h[2]=(__bf16)k0.z; kb.h[3]=(__bf16)k0.w;
      kb.h[4]=(__bf16)k1.x; kb.h[5]=(__bf16)k1.y; kb.h[6]=(__bf16)k1.z; kb.h[7]=(__bf16)k1.w;
      *reinterpret_cast<uint4*>(&Klds[rr * EDIM + se0]) = kb.q;

      const float gv[8] = {g0.x,g0.y,g0.z,g0.w,g1.x,g1.y,g1.z,g1.w};
      const float lv[8] = {l0.x,l0.y,l0.z,l0.w,l1.x,l1.y,l1.z,l1.w};
#pragma unroll
      for (int i = 0; i < 8; ++i) {
        GVt[(se0 + i) * VSTR + rr] = (__bf16)gv[i];
        LVt[(se0 + i) * VSTR + rr] = (__bf16)lv[i];
      }
      if (sc + CHUNK < SEQL) {   // uniform branch: prefetch next chunk
        __builtin_prefetch(kp + (long)CHUNK * ROWSTR, 0, 1);
        __builtin_prefetch(gp + (long)CHUNK * ROWSTR, 0, 1);
        __builtin_prefetch(lp + (long)CHUNK * ROWSTR, 0, 1);
      }
    }
    __syncthreads();

    // ======== scores, transposed: D[key][query] = K_tile x Q^T ===============
    // 4 key tiles of 16; lane holds query m, reg r -> key 16*t + (hi? r+8 : r)
    float sg[4][8], sl[4][8];
    float cmg = -1e30f, cml = -1e30f;
    const float dBase = (float)row - (float)sc - (hiHalf ? 8.0f : 0.0f);
#pragma unroll
    for (int t = 0; t < 4; ++t) {
      const __bf16* kr = &Klds[(m + 16 * t) * EDIM + aeb];
      v16bf ka0 = lds_pair16(kr,      kr + 8);    // e 0..31 runs
      v16bf ka1 = lds_pair16(kr + 32, kr + 40);   // e 32..63 runs
      v8f d = wmma_bf16(ka0, qb0, vzero);
      d = wmma_bf16(ka1, qb1, d);
#pragma unroll
      for (int r = 0; r < 8; ++r) {
        float a  = d[r] * (0.125f * LOG2E);           // score * log2e
        float di = dBase - (float)(16 * t + r);       // row - s  (exact in f32)
        float g  = gA2 * exp2f(di * di * gBn);        // gauss * log2e
        float lo = a + g;
        sg[t][r] = a; sl[t][r] = lo;
        cmg = fmaxf(cmg, a);
        cml = fmaxf(cml, lo);
      }
    }
    // row lives in lanes {m, m+16}: reduce across the halves
    cmg = fmaxf(cmg, __shfl_xor(cmg, 16, 32));
    cml = fmaxf(cml, __shfl_xor(cml, 16, 32));
    const float nmg = fmaxf(mg, cmg);
    const float nml = fmaxf(ml, cml);
    const float ag = exp2f(mg - nmg);
    const float al = exp2f(ml - nml);
    mg = nmg; ml = nml;

    // probabilities -> bf16; layout matches 16-bit A operand (16x32) directly
    BF16x16 pgA, pgB, plA, plB;        // A: s 0..31 (tiles 0,1), B: s 32..63 (tiles 2,3)
    float psg = 0.f, psl = 0.f;
#pragma unroll
    for (int r = 0; r < 8; ++r) {
      float g0 = exp2f(sg[0][r] - nmg), g1 = exp2f(sg[1][r] - nmg);
      float g2 = exp2f(sg[2][r] - nmg), g3 = exp2f(sg[3][r] - nmg);
      float l0 = exp2f(sl[0][r] - nml), l1 = exp2f(sl[1][r] - nml);
      float l2 = exp2f(sl[2][r] - nml), l3 = exp2f(sl[3][r] - nml);
      psg += (g0 + g1) + (g2 + g3);
      psl += (l0 + l1) + (l2 + l3);
      pgA.h[r] = (__bf16)g0; pgA.h[8 + r] = (__bf16)g1;
      pgB.h[r] = (__bf16)g2; pgB.h[8 + r] = (__bf16)g3;
      plA.h[r] = (__bf16)l0; plA.h[8 + r] = (__bf16)l1;
      plB.h[r] = (__bf16)l2; plB.h[8 + r] = (__bf16)l3;
    }
    psg += __shfl_xor(psg, 16, 32);
    psl += __shfl_xor(psl, 16, 32);
    sumg = sumg * ag + psg;
    suml = suml * al + psl;

    // rescale accumulators: reg r holds output row (hi? r+8 : r);
    // its alpha lives in lane (r / r+8) -> broadcast with per-lane-src shfl
#pragma unroll
    for (int r = 0; r < 8; ++r) {
      const int src = hiHalf ? (r + 8) : r;
      float agr = __shfl(ag, src, 32);
      float alr = __shfl(al, src, 32);
      og[0][r] *= agr; og[1][r] *= agr; og[2][r] *= agr; og[3][r] *= agr;
      ol[0][r] *= alr; ol[1][r] *= alr; ol[2][r] *= alr; ol[3][r] *= alr;
    }

    // ======== PV: O[m][e] += P(16x64) x V(64x16 per e-chunk) =================
#pragma unroll
    for (int j = 0; j < 4; ++j) {
      const __bf16* gvp = &GVt[(j * 16 + m) * VSTR + sbh]; // col e, s-contig runs
      const __bf16* lvp = &LVt[(j * 16 + m) * VSTR + sbh];
      v16bf gv0 = lds_pair16(gvp,      gvp + 16);   // s 0..31 half
      v16bf gv1 = lds_pair16(gvp + 32, gvp + 48);   // s 32..63 half
      v16bf lv0 = lds_pair16(lvp,      lvp + 16);
      v16bf lv1 = lds_pair16(lvp + 32, lvp + 48);
      og[j] = wmma_bf16(pgA.v, gv0, og[j]);
      og[j] = wmma_bf16(pgB.v, gv1, og[j]);
      ol[j] = wmma_bf16(plA.v, lv0, ol[j]);
      ol[j] = wmma_bf16(plB.v, lv1, ol[j]);
    }
    __syncthreads();   // protect LDS before next chunk's staging
  }

  // ---------------- epilogue: divide by row sums, store ----------------------
  const float rg = 1.0f / sumg;
  const float rl = 1.0f / suml;
#pragma unroll
  for (int r = 0; r < 8; ++r) {
    const int src = hiHalf ? (r + 8) : r;      // output row index within tile
    float rgr = __shfl(rg, src, 32);
    float rlr = __shfl(rl, src, 32);
    const int orow = rowBase + src;
    const long obase = headBase + (long)orow * ROWSTR + m;  // e col = j*16 + m
#pragma unroll
    for (int j = 0; j < 4; ++j) {
      OutG[obase + j * 16] = og[j][r] * rgr;
      OutL[obase + j * 16] = ol[j][r] * rlr;
    }
  }
}

extern "C" void kernel_launch(void* const* d_in, const int* in_sizes, int n_in,
                              void* d_out, int out_size, void* d_ws, size_t ws_size,
                              hipStream_t stream) {
  (void)in_sizes; (void)n_in; (void)out_size; (void)d_ws; (void)ws_size;
  const float* Q  = (const float*)d_in[0];   // G_queries [B,L,H,E]
  const float* K  = (const float*)d_in[1];   // G_keys    [B,L,H,E]
  const float* GV = (const float*)d_in[2];   // G_values  [B,L,H,E]
  const float* LV = (const float*)d_in[3];   // L_values  [B,L,H,E]
  const float* SG = (const float*)d_in[4];   // sigma     [B,L,H]
  float* outG = (float*)d_out;
  float* outL = outG + (size_t)BATCH * SEQL * NHEAD * EDIM;
  dim3 grid(BATCH * NHEAD * (SEQL / MTILE));  // 512 workgroups of 8 waves
  anomaly_attn_flash_kernel<<<grid, THREADS, 0, stream>>>(Q, K, GV, LV, SG, outG, outL);
}
